// VocoderNetwork_53781580480701
// MI455X (gfx1250) — matrix operand order repaired and checked by reading, not compiled, not run on hardware
//
#include <hip/hip_runtime.h>
#include <hip/hip_bf16.h>
#include <math.h>

typedef __attribute__((ext_vector_type(16))) _Float16 v16h;
typedef __attribute__((ext_vector_type(8)))  float    v8f;

union U16 { v16h h; uint4 q[2]; unsigned u[8]; };

#define SQRT_HALF 0.70710678118f
#define PITCH 520              // 256 f16 channels (512B) + 8B pad -> spreads LDS banks
#define ACT0_OFF 0             // 32  rows * 520 = 16640
#define ACTP_OFF 16640         // 256 rows * 520 = 133120
#define ACTQ_OFF 149760        // 128 rows * 520 = 66560
#define PRE_OFF  216320        // 256 f32
#define LDS_BYTES 217344

// ---------------- prep kernels ----------------

__global__ void k_cond(const float* __restrict__ mgc, const float* __restrict__ cond_w,
                       const float* __restrict__ cond_b, float* __restrict__ cond) {
    int e = blockIdx.x * 256 + threadIdx.x;          // 2*12000 = 24000 = 400*60
    if (e >= 24000) return;
    int f = e / 12000, u = e % 12000;
    const float* m = mgc + f * 60;
    const float* w = cond_w + u * 60;
    float s = cond_b[u];
    for (int k = 0; k < 60; ++k) s += m[k] * w[k];
    cond[e] = tanhf(s);                              // flat == reshape(-1,60)
}

__global__ void k_condproj(const float* __restrict__ cond,
                           const float* __restrict__ cw_in, const float* __restrict__ cb_in,
                           const float* __restrict__ cw_g,  const float* __restrict__ cb_g,
                           float* __restrict__ cp) {
    int e = blockIdx.x * 256 + threadIdx.x;          // 10*2*400*256 = 2048000
    if (e >= 2048000) return;
    int c  = e & 255;
    int n  = (e >> 8) % 400;
    int lw = e / (400 * 256);                        // l*2 + which
    int l  = lw >> 1, which = lw & 1;
    const float* cw = which ? cw_g : cw_in;
    const float* cb = which ? cb_g : cb_in;
    const float* cd = cond + n * 60;
    const float* w  = cw + (l * 256 + c) * 60;
    float s = cb[l * 256 + c];
    for (int k = 0; k < 60; ++k) s += cd[k] * w[k];
    cp[e] = s;                                       // [l][which][n][c]
}

__global__ void k_repack(const float* __restrict__ lw_in, const float* __restrict__ lw_g,
                         const float* __restrict__ lw_r, _Float16* __restrict__ wk) {
    int e = blockIdx.x * 256 + threadIdx.x;          // 9*3*256*512 = 3538944
    if (e >= 3538944) return;
    int k   = e & 511;                               // k = tap*256 + ci
    int co  = (e >> 9) & 255;
    int c3  = (e >> 17) % 3;
    int l   = e / (3 * 256 * 512);
    int tap = k >> 8, ci = k & 255;
    const float* src = (c3 == 0) ? lw_in : (c3 == 1) ? lw_g : lw_r; // [9][256][256][2]
    wk[e] = (_Float16)src[(((l * 256 + co) * 256 + ci) << 1) + tap];
}

// ---------------- fused per-window kernel ----------------

// One 16x16 output tile of a gated conv layer: 3 GEMMs (in/gate/res) over K=512,
// B fragments from LDS activations (row = 2*(t - tin0) + tap, ch-contiguous),
// A fragments from k-major f16 weights in global (L2-resident).
__device__ __forceinline__ void gemm3_tile(
    const char* inb, int tin0, char* outb,
    int mtile, int ntile, int nvalid,
    const _Float16* __restrict__ wk,                 // [3][256][512] this layer
    const float* __restrict__ bin, const float* __restrict__ bg,
    const float* __restrict__ br,
    const float* __restrict__ ci, const float* __restrict__ cg,
    int lane)
{
    int nn = lane & 15, khalf = lane >> 4;
    int m    = mtile * 16 + nn;                      // A row (out channel)
    int ncl  = (nn < nvalid) ? nn : (nvalid - 1);    // clamp cols for narrow layers
    int tcol = ntile * 16 + ncl;                     // B column (output position)
    int tloc = tcol - tin0;

    int kbB[8];
#pragma unroll
    for (int v = 0; v < 8; ++v)
        kbB[v] = (khalf << 3) + (v << 1) + ((v >> 2) << 3); // ISA B 16-bit layout

    v8f accI = {}, accG = {}, accR = {};
    const char* wI = (const char*)(wk + (0 * 256 + m) * 512) + (khalf << 4);
    const char* wG = (const char*)(wk + (1 * 256 + m) * 512) + (khalf << 4);
    const char* wR = (const char*)(wk + (2 * 256 + m) * 512) + (khalf << 4);

#pragma unroll 1
    for (int ks = 0; ks < 16; ++ks) {                // K = 16 * 32; keep rolled (no spills)
        int tap    = ks >> 3;                        // k>=256 -> tap 1 (uniform per ks)
        int rowoff = (2 * tloc + tap) * PITCH;
        int cb2    = (ks * 32 - (tap << 8)) << 1;    // channel byte base
        U16 b;
#pragma unroll
        for (int v = 0; v < 8; ++v)
            b.u[v] = *(const unsigned*)(inb + rowoff + cb2 + (kbB[v] << 1));
        // A fragments: ISA layout == two contiguous 16B spans per khalf group
        U16 aI, aG, aR;
        aI.q[0] = *(const uint4*)(wI + 64 * ks);
        aI.q[1] = *(const uint4*)(wI + 64 * ks + 32);
        aG.q[0] = *(const uint4*)(wG + 64 * ks);
        aG.q[1] = *(const uint4*)(wG + 64 * ks + 32);
        aR.q[0] = *(const uint4*)(wR + 64 * ks);
        aR.q[1] = *(const uint4*)(wR + 64 * ks + 32);
        accI = __builtin_amdgcn_wmma_f32_16x16x32_f16(false, aI.h, false, b.h, (short)0, accI, false, false);
        accG = __builtin_amdgcn_wmma_f32_16x16x32_f16(false, aG.h, false, b.h, (short)0, accG, false, false);
        accR = __builtin_amdgcn_wmma_f32_16x16x32_f16(false, aR.h, false, b.h, (short)0, accR, false, false);
    }

    if (nn < nvalid) {                               // gating + f16 store (post-WMMA divergence ok)
#pragma unroll
        for (int r = 0; r < 8; ++r) {
            int co   = mtile * 16 + r + (khalf << 3);      // ISA C/D layout: M = r + 8*khalf
            float ai = accI[r] + bin[co] + ci[co];
            float ag = accG[r] + bg[co]  + cg[co];
            float ar = accR[r] + br[co];
            float v  = (tanhf(ai) * (1.f / (1.f + expf(-ag))) + ar) * SQRT_HALF;
            *(_Float16*)(outb + tcol * PITCH + (co << 1)) = (_Float16)v;
        }
    }
}

__global__ __launch_bounds__(512)
void vocoder_main(const float* __restrict__ signal,
                  const float* __restrict__ l0_win, const float* __restrict__ l0_bin,
                  const float* __restrict__ l0_wg,  const float* __restrict__ l0_bg,
                  const float* __restrict__ l0_wr,  const float* __restrict__ l0_br,
                  const float* __restrict__ lb_in,  const float* __restrict__ lb_g,
                  const float* __restrict__ lb_r,
                  const float* __restrict__ pre_w,  const float* __restrict__ pre_b,
                  const float* __restrict__ mean_w, const float* __restrict__ mean_b,
                  const float* __restrict__ std_w,  const float* __restrict__ std_b,
                  const float* __restrict__ logit_w,const float* __restrict__ logit_b,
                  const float* __restrict__ condproj,   // [10][2][400][256]
                  const _Float16* __restrict__ wk,      // [9][3][256][512]
                  float* __restrict__ out)              // [3][400][10]
{
    extern __shared__ char lds[];
    char*  ACT0 = lds + ACT0_OFF;
    char*  ACTP = lds + ACTP_OFF;
    char*  ACTQ = lds + ACTQ_OFF;
    float* PRE  = (float*)(lds + PRE_OFF);

    const int n    = blockIdx.x;
    const int tid  = threadIdx.x;
    const int wave = tid >> 5;
    const int lane = tid & 31;

    const float* ci0 = condproj + ((0 * 2 + 0) * 400 + n) * 256;
    const float* cg0 = condproj + ((0 * 2 + 1) * 400 + n) * 256;
    const float* ci1 = condproj + ((1 * 2 + 0) * 400 + n) * 256;
    const float* cg1 = condproj + ((1 * 2 + 1) * 400 + n) * 256;

    // ---- Layer 1, with layer 0 streamed in 32-position tiles ----
#pragma unroll 1
    for (int j = 0; j < 16; ++j) {
#pragma unroll 4
        for (int it = 0; it < 16; ++it) {            // 32 pos * 256 ch = 8192 elems
            int e  = it * 512 + tid;
            int pl = e >> 8, c = e & 255;
            int p  = j * 32 + pl;
            float s0 = signal[n + 2 * p], s1 = signal[n + 2 * p + 1];
            float aI = l0_win[c * 2 + 0] * s0 + l0_win[c * 2 + 1] * s1 + l0_bin[c] + ci0[c];
            float aG = l0_wg [c * 2 + 0] * s0 + l0_wg [c * 2 + 1] * s1 + l0_bg [c] + cg0[c];
            float aR = l0_wr [c * 2 + 0] * s0 + l0_wr [c * 2 + 1] * s1 + l0_br [c];
            float v  = (tanhf(aI) * (1.f / (1.f + expf(-aG))) + aR) * SQRT_HALF;
            *(_Float16*)(ACT0 + pl * PITCH + (c << 1)) = (_Float16)v;
        }
        __syncthreads();
        gemm3_tile(ACT0, j * 16, ACTP, wave, j, 16,
                   wk /* layer 1 -> wk[0] */,
                   lb_in, lb_g, lb_r, ci1, cg1, lane);
        __syncthreads();
    }

    // ---- Layers 2..9, LDS ping-pong ----
    char* inb = ACTP;
    char* outb = ACTQ;
#pragma unroll 1
    for (int l = 2; l <= 9; ++l) {
        int Nl     = 512 >> l;
        int ntiles = (Nl + 15) >> 4;
        int nvalid = (Nl < 16) ? Nl : 16;
        const float* ciL = condproj + ((l * 2 + 0) * 400 + n) * 256;
        const float* cgL = condproj + ((l * 2 + 1) * 400 + n) * 256;
        const _Float16* wkl = wk + (size_t)(l - 1) * 3 * 256 * 512;
        const float* bi  = lb_in + (l - 1) * 256;
        const float* bg_ = lb_g  + (l - 1) * 256;
        const float* br_ = lb_r  + (l - 1) * 256;
        int total = 16 * ntiles;                     // multiple of 16 -> wave-uniform
#pragma unroll 1
        for (int tile = wave; tile < total; tile += 16)
            gemm3_tile(inb, 0, outb, tile & 15, tile >> 4, nvalid,
                       wkl, bi, bg_, br_, ciL, cgL, lane);
        __syncthreads();
        char* t = inb; inb = outb; outb = t;
    }
    // act9 (1 position x 256 ch, f16) now at row 0 of inb

    // ---- head: pre = relu(act9 @ pre_w.T + pre_b) ----
    if (tid < 256) {
        float s = pre_b[tid];
        const float* pw = pre_w + tid * 256;
        for (int k = 0; k < 256; ++k)
            s += (float)(*(const _Float16*)(inb + (k << 1))) * pw[k];
        PRE[tid] = s > 0.f ? s : 0.f;
    }
    __syncthreads();
    if (tid < 30) {
        int which = tid / 10, i = tid % 10;
        const float* w = (which == 0) ? mean_w : (which == 1) ? std_w : logit_w;
        const float* b = (which == 0) ? mean_b : (which == 1) ? std_b : logit_b;
        float s = b[i];
        for (int k = 0; k < 256; ++k) s += PRE[k] * w[i * 256 + k];
        out[which * 4000 + n * 10 + i] = s;          // mean | stdev | logits, each [400][10]
    }
}

// ---------------- launch ----------------

extern "C" void kernel_launch(void* const* d_in, const int* in_sizes, int n_in,
                              void* d_out, int out_size, void* d_ws, size_t ws_size,
                              hipStream_t stream) {
    const float* signal  = (const float*)d_in[0];
    const float* mgc     = (const float*)d_in[1];
    const float* cond_w  = (const float*)d_in[2];
    const float* cond_b  = (const float*)d_in[3];
    const float* l0_win  = (const float*)d_in[4];
    const float* l0_bin  = (const float*)d_in[5];
    const float* l0_wg   = (const float*)d_in[6];
    const float* l0_bg   = (const float*)d_in[7];
    const float* l0_wr   = (const float*)d_in[8];
    const float* l0_br   = (const float*)d_in[9];
    const float* lw_in   = (const float*)d_in[10];
    const float* lb_in   = (const float*)d_in[11];
    const float* lw_g    = (const float*)d_in[12];
    const float* lb_g    = (const float*)d_in[13];
    const float* lw_r    = (const float*)d_in[14];
    const float* lb_r    = (const float*)d_in[15];
    const float* cw_in   = (const float*)d_in[16];
    const float* cb_in   = (const float*)d_in[17];
    const float* cw_g    = (const float*)d_in[18];
    const float* cb_g    = (const float*)d_in[19];
    const float* pre_w   = (const float*)d_in[20];
    const float* pre_b   = (const float*)d_in[21];
    const float* mean_w  = (const float*)d_in[22];
    const float* mean_b  = (const float*)d_in[23];
    const float* std_w   = (const float*)d_in[24];
    const float* std_b   = (const float*)d_in[25];
    const float* logit_w = (const float*)d_in[26];
    const float* logit_b = (const float*)d_in[27];

    char* ws = (char*)d_ws;
    float*    condbuf = (float*)(ws);                 // 24000 f32
    float*    cp      = (float*)(ws + 98304);         // 10*2*400*256 f32 = 8192000 B
    _Float16* wk      = (_Float16*)(ws + 8290304);    // 9*3*256*512 f16 = 7077888 B

    k_cond    <<<(24000   + 255) / 256, 256, 0, stream>>>(mgc, cond_w, cond_b, condbuf);
    k_condproj<<<(2048000 + 255) / 256, 256, 0, stream>>>(condbuf, cw_in, cb_in, cw_g, cb_g, cp);
    k_repack  <<<(3538944 + 255) / 256, 256, 0, stream>>>(lw_in, lw_g, lw_r, wk);

    vocoder_main<<<400, 512, LDS_BYTES, stream>>>(
        signal, l0_win, l0_bin, l0_wg, l0_bg, l0_wr, l0_br,
        lb_in, lb_g, lb_r, pre_w, pre_b, mean_w, mean_b, std_w, std_b,
        logit_w, logit_b, cp, wk, (float*)d_out);
}
